// LSTMTagger_25262997635706
// MI455X (gfx1250) — compile-verified, matrix operand order checked by reading, not run on hardware
//
#include <hip/hip_runtime.h>
#include <hip/hip_bf16.h>
#include <math.h>

// ---------------------------------------------------------------------------
// LSTM tagger for MI455X (gfx1250, wave32, WMMA + TDM/async-LDS).
//   K0 prep      : f32->f16 weight swizzle into WMMA B-layout, bias folding.
//   K1 char LSTM : 1024 blocks x 1 wave, 16 words/block, 12 steps,
//                  3x v_wmma_f32_16x16x32_f16 per 16-col gate tile;
//                  char-emb table staged via global_load_async_to_lds_b128.
//   K2 word LSTM : 32 persistent blocks, weights LDS-resident via
//                  tensor_load_to_lds (TDM) + s_wait_tensorcnt, per-step
//                  release counter + s_cluster_barrier, double-buffered h.
//   K3 head      : WMMA GEMM vs lin_W^T (padded 50->64) + log_softmax;
//                  h rows staged via async-to-LDS.
// Workspace need: ~20.1 MB.
// ---------------------------------------------------------------------------

typedef __attribute__((ext_vector_type(16))) _Float16 v16h;
typedef __attribute__((ext_vector_type(8)))  float    v8f;
typedef unsigned int uint32x4 __attribute__((ext_vector_type(4)));
typedef int          int32x8  __attribute__((ext_vector_type(8)));
typedef int          int32x4  __attribute__((ext_vector_type(4)));

#define S_  16384
#define L_  12
#define E_  128
#define NC_ 64
#define DC_ 32
#define HC_ 64
#define H_  256
#define NT_ 50

// workspace byte offsets
#define OFF_CB    0u                         // 256 f32  (c_bih + c_bhh)
#define OFF_CW    1024u                      // 16 tiles * 512 f16 (c_Wih^T)
#define OFF_CU    17408u                     // 32 tiles * 512 f16 (c_Whh^T)
#define OFF_LW    50176u                     // 32 tiles * 512 f16 (lin_W^T pad)
#define OFF_HBUF  82944u                     // 2 * 256 f32 (h double buffer)
#define OFF_CNT   84992u                     // u32 step counter
#define OFF_CF    85248u                     // S*64 f32 char features
#define OFF_HS    (85248u + (unsigned)S_*HC_*4u)  // S*256 f32 word-LSTM h's

#if defined(__has_builtin)
# if __has_builtin(__builtin_amdgcn_tensor_load_to_lds)
#  define HAVE_TDM 1
# endif
#endif
#if defined(__clang_major__) && (__clang_major__ >= 23)
# define TDM_SIX_ARGS 1
#endif

#if defined(__has_builtin) && __has_builtin(__builtin_amdgcn_s_wait_tensorcnt)
# define WAIT_TENSOR() __builtin_amdgcn_s_wait_tensorcnt(0)
#else
# define WAIT_TENSOR() asm volatile("s_wait_tensorcnt 0x0" ::: "memory")
#endif
#define WAIT_ASYNC() asm volatile("s_wait_asynccnt 0x0" ::: "memory")

__device__ __forceinline__ float sigmoidf_(float x) {
  return 1.0f / (1.0f + __expf(-x));
}

// A-matrix (16x32 f16) K index for half j, lane-group g (wave32 layout):
// lanes 0-15 hold M=0..15 / K 0-7,16-23 ; lanes 16-31 hold K 8-15,24-31.
__device__ __forceinline__ int a_k(int j, int g) {
  return ((j >> 3) << 4) + (g << 3) + (j & 7);
}

// Per-lane async 16B copy global -> LDS (ASYNCcnt tracked).
__device__ __forceinline__ void async_copy_b128(void* lds_dst,
                                                const void* gsrc) {
  unsigned           lds = (unsigned)(uintptr_t)lds_dst;
  unsigned long long ga  = (unsigned long long)(uintptr_t)gsrc;
  asm volatile("global_load_async_to_lds_b128 %0, %1, off"
               :: "v"(lds), "v"(ga) : "memory");
}

#if defined(HAVE_TDM)
// Issue a 1-D TDM load: nelems f32 from gptr -> LDS byte address lds_addr.
// D# group0: count=1, lds_addr, 57-bit global addr, type=2 ("image").
// D# group1: data_size=4B, tensor_dim0=tile_dim0=nelems, dim0_stride=nelems.
__device__ __forceinline__ void tdm_load_1d(unsigned lds_addr,
                                            const void* gptr,
                                            unsigned nelems) {
  unsigned long long ga = (unsigned long long)(uintptr_t)gptr;
  uint32x4 g0 = { 1u,                                        // count=1
                  lds_addr,                                  // lds_addr
                  (unsigned)(ga & 0xffffffffu),              // gaddr lo
                  (unsigned)((ga >> 32) & 0x01ffffffu) | (2u << 30) }; // type=2
  int32x8 g1 = { (int)(2u << 16),                            // data_size=4B
                 (int)((nelems & 0xffffu) << 16),            // tensor_dim0 lo16
                 (int)((nelems >> 16) | (1u << 16)),         // dim0 hi | dim1=1
                 (int)((nelems & 0xffffu) << 16),            // tile_dim0
                 0,                                          // tile_dim1/2 = 0
                 (int)nelems,                                // dim0_stride lo32
                 0, 0 };
  int32x4 z4 = { 0, 0, 0, 0 };
#if defined(TDM_SIX_ARGS)
  int32x8 z8 = { 0, 0, 0, 0, 0, 0, 0, 0 };
  __builtin_amdgcn_tensor_load_to_lds(g0, g1, z4, z4, z8, 0);
#else
  __builtin_amdgcn_tensor_load_to_lds(g0, g1, z4, z4, 0);
#endif
}
#endif

// ---------------------------------------------------------------------------
// K0: prepare swizzled f16 B-tiles + folded biases + sync state.
// B-tile storage: tile*512 + lane*16 + j halves; element = W[n_glob][k_glob]
// with n = lane&15, g = lane>>4, k = kt*32 + a_k(j,g)  (mirror of A layout).
// ---------------------------------------------------------------------------
__global__ void k_prep(const float* __restrict__ c_Wih,
                       const float* __restrict__ c_Whh,
                       const float* __restrict__ c_bih,
                       const float* __restrict__ c_bhh,
                       const float* __restrict__ lin_W,
                       float* __restrict__ cb,
                       _Float16* __restrict__ cW,
                       _Float16* __restrict__ cU,
                       _Float16* __restrict__ lW,
                       float* __restrict__ hbuf,
                       unsigned* __restrict__ cnt) {
  int gid    = blockIdx.x * blockDim.x + threadIdx.x;
  int stride = gridDim.x * blockDim.x;

  // c_Wih^T : 1 k-tile x 16 n-tiles
  for (int idx = gid; idx < 16 * 512; idx += stride) {
    int tile = idx >> 9, w = idx & 511;
    int lane = w >> 4, j = w & 15;
    int n = lane & 15, g = lane >> 4;
    int k = a_k(j, g);
    cW[idx] = (_Float16)c_Wih[(tile * 16 + n) * DC_ + k];
  }
  // c_Whh^T : tiles = nt*2 + kt
  for (int idx = gid; idx < 32 * 512; idx += stride) {
    int tile = idx >> 9, w = idx & 511;
    int lane = w >> 4, j = w & 15;
    int n = lane & 15, g = lane >> 4;
    int nt = tile >> 1, kt = tile & 1;
    int k = kt * 32 + a_k(j, g);
    cU[idx] = (_Float16)c_Whh[(nt * 16 + n) * HC_ + k];
  }
  // lin_W^T padded to 64 cols : tiles = nt*8 + kt
  for (int idx = gid; idx < 32 * 512; idx += stride) {
    int tile = idx >> 9, w = idx & 511;
    int lane = w >> 4, j = w & 15;
    int n = lane & 15, g = lane >> 4;
    int nt = tile >> 3, kt = tile & 7;
    int k = kt * 32 + a_k(j, g);
    int ng = nt * 16 + n;
    lW[idx] = (ng < NT_) ? (_Float16)lin_W[ng * H_ + k] : (_Float16)0.0f;
  }
  for (int idx = gid; idx < 4 * HC_; idx += stride)
    cb[idx] = c_bih[idx] + c_bhh[idx];
  for (int idx = gid; idx < 2 * H_; idx += stride)
    hbuf[idx] = 0.0f;
  if (gid == 0) *cnt = 0u;
}

// ---------------------------------------------------------------------------
// K1: char-level LSTM, 16 words per single-wave block, WMMA gate GEMMs.
// ---------------------------------------------------------------------------
__global__ __launch_bounds__(32) void k_char_lstm(
    const int*   __restrict__ char_idx,
    const int*   __restrict__ char_lens,
    const float* __restrict__ char_emb,
    const float* __restrict__ cb,
    const _Float16* __restrict__ cW,
    const _Float16* __restrict__ cU,
    float* __restrict__ cfeat) {
  __shared__ float ce_l[NC_ * DC_];       // whole char embedding table (8KB)
  __shared__ int   cidx_l[16 * L_];
  __shared__ int   len_l[16];
  __shared__ float h_l[16 * HC_];
  __shared__ float c_l[16 * HC_];
  __shared__ float pre_l[16 * 4 * HC_];   // 16 x 256 gate pre-activations

  const int tid = threadIdx.x;
  const int w0  = blockIdx.x * 16;
  const int m   = tid & 15;   // M row for A / N col for C-store
  const int g   = tid >> 4;   // lane group

  // Async-stage the f32 char embedding table: 512 x b128 chunks, 32 lanes.
  for (int it = 0; it < 16; ++it) {
    const int c4 = (it * 32 + tid) * 4;   // f32 index, 16B granularity
    async_copy_b128(&ce_l[c4], char_emb + c4);
  }
  for (int idx = tid; idx < 16 * L_; idx += 32)
    cidx_l[idx] = char_idx[(size_t)(w0 + idx / L_) * L_ + (idx % L_)];
  if (tid < 16) len_l[tid] = char_lens[w0 + tid];
  for (int idx = tid; idx < 16 * HC_; idx += 32) {
    h_l[idx] = 0.0f;
    c_l[idx] = 0.0f;
  }
  WAIT_ASYNC();
  __syncthreads();

  for (int t = 0; t < L_; ++t) {
    // Build A matrices: x_t (16x32) and h (16x64 -> two 16x32 slices).
    v16h ax, ah0, ah1;
    const int ci = cidx_l[m * L_ + t];
#pragma unroll
    for (int j = 0; j < 16; ++j) {
      const int K = a_k(j, g);
      ax[j]  = (_Float16)ce_l[ci * DC_ + K];
      ah0[j] = (_Float16)h_l[m * HC_ + K];
      ah1[j] = (_Float16)h_l[m * HC_ + 32 + K];
    }
    // 16 N-tiles of the 256 gate columns, 3 WMMAs each (x, h-lo, h-hi).
    for (int nt = 0; nt < 16; ++nt) {
      const float bv = cb[nt * 16 + m];
      v8f acc;
#pragma unroll
      for (int r = 0; r < 8; ++r) acc[r] = bv;
      v16h bx = *(const v16h*)(cW + (size_t)nt * 512 + tid * 16);
      acc = __builtin_amdgcn_wmma_f32_16x16x32_f16(false, ax, false, bx,
                                                   (short)0, acc, false, false);
      v16h bh0 = *(const v16h*)(cU + (size_t)(nt * 2 + 0) * 512 + tid * 16);
      acc = __builtin_amdgcn_wmma_f32_16x16x32_f16(false, ah0, false, bh0,
                                                   (short)0, acc, false, false);
      v16h bh1 = *(const v16h*)(cU + (size_t)(nt * 2 + 1) * 512 + tid * 16);
      acc = __builtin_amdgcn_wmma_f32_16x16x32_f16(false, ah1, false, bh1,
                                                   (short)0, acc, false, false);
      // C/D layout: VGPR r, lanes 0-15 -> M=r, lanes 16-31 -> M=r+8; N=lane&15
#pragma unroll
      for (int r = 0; r < 8; ++r)
        pre_l[(r + 8 * g) * (4 * HC_) + nt * 16 + m] = acc[r];
    }
    __syncthreads();
    // Gate nonlinearity + masked state update.
    for (int idx = tid; idx < 16 * HC_; idx += 32) {
      const int mm = idx >> 6, j = idx & 63;
      if (t < len_l[mm]) {
        const float* p = &pre_l[mm * (4 * HC_)];
        float pi = p[j], pf = p[64 + j], pg = p[128 + j], po = p[192 + j];
        float cn = sigmoidf_(pf) * c_l[idx] + sigmoidf_(pi) * tanhf(pg);
        c_l[idx] = cn;
        h_l[idx] = sigmoidf_(po) * tanhf(cn);
      }
    }
    __syncthreads();
  }
  for (int idx = tid; idx < 16 * HC_; idx += 32)
    cfeat[(size_t)w0 * HC_ + idx] = h_l[idx];
}

// ---------------------------------------------------------------------------
// K2: word-level LSTM, 32 persistent blocks each owning 8 h-elements
// (rows j, 256+j, 512+j, 768+j of the gate matrix) with weights in LDS.
// ---------------------------------------------------------------------------
__global__ __launch_bounds__(256) void k_word_lstm(
    const int*   __restrict__ x,
    const float* __restrict__ word_emb,
    const float* __restrict__ cfeat,
    const float* __restrict__ w_Wih,
    const float* __restrict__ w_Whh,
    const float* __restrict__ w_bih,
    const float* __restrict__ w_bhh,
    float* __restrict__ hbuf,
    unsigned* __restrict__ cnt,
    float* __restrict__ hs) {
  __shared__ float wih_l[32 * 192];  // 24 KB
  __shared__ float whh_l[32 * 256];  // 32 KB
  __shared__ float vec_l[448];       // [x_t(192) | h(256)]
  __shared__ float pre_l[32];
  __shared__ float rb_l[32];
  __shared__ float cst[8];

  const int tid = threadIdx.x;
  const int b   = blockIdx.x;

  // Stage this block's 32 gate rows into LDS (resident for all 16384 steps).
#if defined(HAVE_TDM)
  if (tid == 0) {
    for (int gate = 0; gate < 4; ++gate) {
      // rows gate*256 + b*8 .. +8 are contiguous -> 1-D TDM tiles.
      tdm_load_1d((unsigned)(uintptr_t)&wih_l[gate * 8 * 192],
                  w_Wih + ((size_t)gate * H_ + b * 8) * 192, 8 * 192);
      tdm_load_1d((unsigned)(uintptr_t)&whh_l[gate * 8 * 256],
                  w_Whh + ((size_t)gate * H_ + b * 8) * 256, 8 * 256);
    }
    WAIT_TENSOR();
  }
#else
  for (int idx = tid; idx < 32 * 192; idx += 256) {
    int r = idx / 192, k = idx % 192;
    int grow = (r >> 3) * H_ + b * 8 + (r & 7);
    wih_l[idx] = w_Wih[(size_t)grow * 192 + k];
  }
  for (int idx = tid; idx < 32 * 256; idx += 256) {
    int r = idx >> 8, k = idx & 255;
    int grow = (r >> 3) * H_ + b * 8 + (r & 7);
    whh_l[idx] = w_Whh[(size_t)grow * H_ + k];
  }
#endif
  if (tid < 32) {
    int grow = (tid >> 3) * H_ + b * 8 + (tid & 7);
    rb_l[tid] = w_bih[grow] + w_bhh[grow];
  }
  if (tid < 8) cst[tid] = 0.0f;
  __syncthreads();

  volatile unsigned* vcnt = cnt;

  for (int t = 0; t < S_; ++t) {
    // Gather step input into LDS: word emb + char feature + broadcast h.
    if (tid < 192) {
      vec_l[tid] = (tid < E_) ? word_emb[(size_t)x[t] * E_ + tid]
                              : cfeat[(size_t)t * HC_ + (tid - E_)];
    }
    vec_l[192 + tid] = hbuf[(t & 1) * H_ + tid];
    if (tid == 0 && t + 1 < S_)
      __builtin_prefetch(word_emb + (size_t)x[t + 1] * E_, 0, 1);
    __syncthreads();

    // 8 lanes per gate row: split the 448-long dot product.
    const int r = tid >> 3, c = tid & 7;
    float sum = 0.0f;
    const float* wr = &wih_l[r * 192 + c * 24];
    const float* vx = &vec_l[c * 24];
#pragma unroll
    for (int k = 0; k < 24; ++k) sum += wr[k] * vx[k];
    const float* ur = &whh_l[r * 256 + c * 32];
    const float* vh = &vec_l[192 + c * 32];
#pragma unroll
    for (int k = 0; k < 32; ++k) sum += ur[k] * vh[k];
    sum += __shfl_xor(sum, 4, 8);
    sum += __shfl_xor(sum, 2, 8);
    sum += __shfl_xor(sum, 1, 8);
    if (c == 0) pre_l[r] = sum + rb_l[r];
    __syncthreads();

    if (tid < 8) {
      float pi = pre_l[tid], pf = pre_l[8 + tid];
      float pg = pre_l[16 + tid], po = pre_l[24 + tid];
      float cn = sigmoidf_(pf) * cst[tid] + sigmoidf_(pi) * tanhf(pg);
      cst[tid] = cn;
      float hn = sigmoidf_(po) * tanhf(cn);
      const int gj = b * 8 + tid;
      hbuf[((t + 1) & 1) * H_ + gj] = hn;   // broadcast for next step
      hs[(size_t)t * H_ + gj] = hn;         // record for the head GEMM
    }
    __threadfence();
    __syncthreads();
    if (tid == 0) atomicAdd(cnt, 1u);
    __builtin_amdgcn_s_cluster_barrier();   // NOP unless cluster-dispatched
    if (tid == 0) {
      const unsigned target = 32u * (unsigned)(t + 1);
      while (*vcnt < target) __builtin_amdgcn_s_sleep(1);
    }
    __syncthreads();
    __threadfence();
  }
}

// ---------------------------------------------------------------------------
// K3: logits = hs @ lin_W^T (+b) via WMMA (N padded 50->64), then log_softmax.
// ---------------------------------------------------------------------------
__global__ __launch_bounds__(32) void k_head(
    const float* __restrict__ hs,
    const _Float16* __restrict__ lW,
    const float* __restrict__ lin_b,
    float* __restrict__ out) {
  __shared__ float hs_l[16 * H_];   // 16 rows of h (16KB)
  __shared__ float lg_l[16 * 64];   // padded logits

  const int tid = threadIdx.x;
  const int w0  = blockIdx.x * 16;
  const int m   = tid & 15, g = tid >> 4;

  // Async-stage 16 h rows: 1024 x b128 chunks across 32 lanes.
  for (int it = 0; it < 32; ++it) {
    const int c4 = (it * 32 + tid) * 4;   // f32 index, 16B granularity
    async_copy_b128(&hs_l[c4], hs + (size_t)w0 * H_ + c4);
  }
  WAIT_ASYNC();
  __syncthreads();

  for (int nt = 0; nt < 4; ++nt) {
    const int ng = nt * 16 + m;
    const float bias = (ng < NT_) ? lin_b[ng] : 0.0f;
    v8f acc;
#pragma unroll
    for (int r = 0; r < 8; ++r) acc[r] = bias;
#pragma unroll
    for (int kt = 0; kt < 8; ++kt) {
      v16h a;
#pragma unroll
      for (int j = 0; j < 16; ++j)
        a[j] = (_Float16)hs_l[m * H_ + kt * 32 + a_k(j, g)];
      v16h bmat = *(const v16h*)(lW + (size_t)(nt * 8 + kt) * 512 + tid * 16);
      acc = __builtin_amdgcn_wmma_f32_16x16x32_f16(false, a, false, bmat,
                                                   (short)0, acc, false, false);
    }
#pragma unroll
    for (int r = 0; r < 8; ++r)
      lg_l[(r + 8 * g) * 64 + nt * 16 + m] = acc[r];
  }
  __syncthreads();

  if (tid < 16) {
    const float* row = &lg_l[tid * 64];
    float mx = -1e30f;
    for (int j = 0; j < NT_; ++j) mx = fmaxf(mx, row[j]);
    float s = 0.0f;
    for (int j = 0; j < NT_; ++j) s += __expf(row[j] - mx);
    const float lse = mx + __logf(s);
    float* o = &out[(size_t)(w0 + tid) * NT_];
    for (int j = 0; j < NT_; ++j) o[j] = row[j] - lse;
  }
}

// ---------------------------------------------------------------------------
extern "C" void kernel_launch(void* const* d_in, const int* in_sizes, int n_in,
                              void* d_out, int out_size, void* d_ws,
                              size_t ws_size, hipStream_t stream) {
  const int*   x         = (const int*)d_in[0];
  const int*   char_idx  = (const int*)d_in[1];
  const int*   char_lens = (const int*)d_in[2];
  const float* word_emb  = (const float*)d_in[3];
  const float* char_emb  = (const float*)d_in[4];
  const float* c_Wih     = (const float*)d_in[5];
  const float* c_Whh     = (const float*)d_in[6];
  const float* c_bih     = (const float*)d_in[7];
  const float* c_bhh     = (const float*)d_in[8];
  const float* w_Wih     = (const float*)d_in[9];
  const float* w_Whh     = (const float*)d_in[10];
  const float* w_bih     = (const float*)d_in[11];
  const float* w_bhh     = (const float*)d_in[12];
  const float* lin_W     = (const float*)d_in[13];
  const float* lin_b     = (const float*)d_in[14];

  char* ws = (char*)d_ws;
  float*    cb    = (float*)(ws + OFF_CB);
  _Float16* cW    = (_Float16*)(ws + OFF_CW);
  _Float16* cU    = (_Float16*)(ws + OFF_CU);
  _Float16* lW    = (_Float16*)(ws + OFF_LW);
  float*    hbuf  = (float*)(ws + OFF_HBUF);
  unsigned* cnt   = (unsigned*)(ws + OFF_CNT);
  float*    cfeat = (float*)(ws + OFF_CF);
  float*    hs    = (float*)(ws + OFF_HS);
  float*    out   = (float*)d_out;

  k_prep<<<64, 256, 0, stream>>>(c_Wih, c_Whh, c_bih, c_bhh, lin_W,
                                 cb, cW, cU, lW, hbuf, cnt);
  k_char_lstm<<<S_ / 16, 32, 0, stream>>>(char_idx, char_lens, char_emb,
                                          cb, cW, cU, cfeat);
  k_word_lstm<<<32, 256, 0, stream>>>(x, word_emb, cfeat, w_Wih, w_Whh,
                                      w_bih, w_bhh, hbuf, cnt, hs);
  k_head<<<S_ / 16, 32, 0, stream>>>(hs, lW, lin_b, out);
}
